// SparseDepthEstimator_9749575762599
// MI455X (gfx1250) — matrix operand order repaired and checked by reading, not compiled, not run on hardware
//
#include <hip/hip_runtime.h>
#include <math.h>

typedef __attribute__((ext_vector_type(16))) _Float16 v16h;
typedef __attribute__((ext_vector_type(8)))  float    v8f;

#define BATCH 16
#define CIN   128
#define HH    192
#define WW    192
#define OC    64
#define HW    (HH*WW)
#define KPTS  512
#define CP    40       // padded channel stride (halves) in LDS: 80B -> 16B aligned, bank-rotating
#define WPACK 73728    // packed weight halves per batch (4 mtiles * 9 taps * 4 chunks * 512)

// ---------------------------------------------------------------------------
// 1) pool[b][c] = mean over HxW
// ---------------------------------------------------------------------------
__global__ void pool_kernel(const float* __restrict__ feat, float* __restrict__ pool) {
    int bc = blockIdx.x;                       // b*CIN + c
    const float* p = feat + (size_t)bc * HW;
    float s = 0.f;
    for (int i = threadIdx.x; i < HW; i += 256) s += p[i];
    __shared__ float red[256];
    red[threadIdx.x] = s; __syncthreads();
    for (int o = 128; o > 0; o >>= 1) {
        if (threadIdx.x < o) red[threadIdx.x] += red[threadIdx.x + o];
        __syncthreads();
    }
    if (threadIdx.x == 0) pool[bc] = red[0] * (1.0f / (float)HW);
}

// ---------------------------------------------------------------------------
// 2) attn = sigmoid(pool @ attn_w^T + attn_b)   (16x128 @ 128x128 -- tiny)
// ---------------------------------------------------------------------------
__global__ void attn_kernel(const float* __restrict__ pool, const float* __restrict__ aw,
                            const float* __restrict__ ab, float* __restrict__ attn) {
    int gid = blockIdx.x * 256 + threadIdx.x;  // 0..2047
    int b = gid >> 7, c = gid & 127;
    float s = ab[c];
    const float* pr = pool + b * CIN;
    const float* wr = aw + c * CIN;
    #pragma unroll 4
    for (int k = 0; k < CIN; ++k) s += pr[k] * wr[k];
    attn[gid] = 1.0f / (1.0f + expf(-s));
}

// ---------------------------------------------------------------------------
// 3) pack conv1_w (64,128,3,3) f32 -> f16, with per-batch channel attention
//    folded in (conv(x*attn_c, W) == conv(x, W*attn_c)), in exact WMMA
//    A-fragment lane order: [b][m_tile(4)][tap(9)][c_chunk(4)][lane(32)][16 halves]
//    A(16x32): lane m=lane&15, hi=lane>>4; elem (i,j): K = 16*(i>>2)+8*hi+2*(i&3)+j
// ---------------------------------------------------------------------------
__global__ void pack_w_kernel(const float* __restrict__ w1, const float* __restrict__ attn,
                              _Float16* __restrict__ wp) {
    int b   = blockIdx.x / 288;
    int gid = (blockIdx.x % 288) * 256 + threadIdx.x;  // < 73728
    int mt  = gid / 18432;  int r1 = gid % 18432;
    int tap = r1 / 2048;    int r2 = r1 % 2048;
    int cch = r2 / 512;     int r3 = r2 % 512;
    int lane = r3 >> 4;     int hx = r3 & 15;
    int i = hx >> 1, j = hx & 1, hi = lane >> 4, m = lane & 15;
    int klocal = 16 * (i >> 2) + 8 * hi + 2 * (i & 3) + j;
    int o = mt * 16 + m, c = cch * 32 + klocal;
    int kh = tap / 3, kw = tap % 3;
    float v = w1[((o * CIN + c) * 3 + kh) * 3 + kw] * attn[b * CIN + c];
    wp[(size_t)b * WPACK + gid] = (_Float16)v;
}

// ---------------------------------------------------------------------------
// 4) Fused: feat -> f16 LDS stage -> implicit-GEMM 3x3 conv via WMMA
//    (attn pre-folded into weights) -> +bias -> exact GELU -> 1x1 conv2 ->
//    depth / conf-logit row outputs.
//    Grid: B*H blocks x 256 threads (8 waves: 4 M-tiles x 2 N-groups).
// ---------------------------------------------------------------------------
__global__ void __launch_bounds__(256) conv_fused_kernel(
    const float* __restrict__ feat, const _Float16* __restrict__ wp,
    const float* __restrict__ b1, const float* __restrict__ w2,
    const float* __restrict__ b2,
    float* __restrict__ depth_out, float* __restrict__ logit_out) {

    int blk = blockIdx.x;
    int b = blk / HH, h = blk % HH;
    int tid = threadIdx.x;
    int lane = tid & 31, wave = tid >> 5;
    int mtile = wave & 3, ngrp = wave >> 2;
    int hi = lane >> 4, n = lane & 15;

    const _Float16* wpb = wp + (size_t)b * WPACK;

    // 3 input rows x 194 cols x 32 channels, channel-contiguous, padded stride
    __shared__ __align__(16) _Float16 xs[3 * 194 * CP];        // 46,560 B
    __shared__ float sacc[2][4][WW];                           //  6,144 B

    v8f vzero = {};
    v8f acc[6];
    #pragma unroll
    for (int t = 0; t < 6; ++t) acc[t] = vzero;

    #pragma unroll 1
    for (int cc = 0; cc < 4; ++cc) {                           // K chunks of 32 channels
        __syncthreads();                                       // prev compute done
        const int cbase = cc * 32;
        // stage: read f32 feature, convert to f16 (attn already in weights)
        for (int idx = tid; idx < 3 * 194 * 32; idx += 256) {
            int c = idx / (3 * 194); int rem = idx % (3 * 194);
            int r = rem / 194, col = rem % 194;
            int hh2 = h - 1 + r, ww2 = col - 1;
            float v = 0.f;
            if (hh2 >= 0 && hh2 < HH && ww2 >= 0 && ww2 < WW)
                v = feat[(((size_t)b * CIN + cbase + c) * HH + hh2) * WW + ww2];
            xs[(r * 194 + col) * CP + c] = (_Float16)v;
        }
        __syncthreads();
        // prefetch next chunk's rows into cache while the WMMA loop runs
        if (cc < 3) {
            int seg = tid;                                     // 96 segs: 32 ch x 3 rows
            if (seg < 96) {
                int c = seg / 3, r = seg % 3;
                int hh2 = h - 1 + r;
                if (hh2 >= 0 && hh2 < HH) {
                    const float* pf = &feat[(((size_t)b * CIN + cbase + 32 + c) * HH + hh2) * WW];
                    __builtin_prefetch(pf, 0, 0);
                    __builtin_prefetch(pf + 96, 0, 0);
                }
            }
        }
        #pragma unroll 1
        for (int tap = 0; tap < 9; ++tap) {
            int kh = tap / 3, kw = tap - kh * 3;
            // A fragment: one 32B coalesced load per lane (pre-packed layout)
            const _Float16* ap = wpb + (((mtile * 9 + tap) * 4 + cc) << 9) + (lane << 4);
            v16h a = *(const v16h*)ap;
            #pragma unroll
            for (int nt = 0; nt < 6; ++nt) {
                int w0 = (ngrp * 6 + nt) * 16;
                // B fragment: K = 8*hi..8*hi+7 and 16+8*hi..16+8*hi+7 = 2 aligned b128 LDS loads
                union { uint4 q[2]; v16h hv; } bf;
                const _Float16* bp = &xs[(kh * 194 + w0 + n + kw) * CP + 8 * hi];
                bf.q[0] = *(const uint4*)bp;
                bf.q[1] = *(const uint4*)(bp + 16);
                acc[nt] = __builtin_amdgcn_wmma_f32_16x16x32_f16(
                    false, a, false, bf.hv, (short)0, acc[nt], false, false);
            }
        }
    }

    // Epilogue: bias + exact GELU + 1x1 conv2 reduction over 64 channels
    __syncthreads();
    float c2b0 = b2[0], c2b1 = b2[1];
    #pragma unroll
    for (int nt = 0; nt < 6; ++nt) {
        float dsum = 0.f, csum = 0.f;
        #pragma unroll
        for (int v = 0; v < 8; ++v) {
            int ch = mtile * 16 + 8 * hi + v;                  // C-fragment M mapping
            float yv = acc[nt][v] + b1[ch];
            float g  = 0.5f * yv * (1.0f + erff(yv * 0.70710678118654752f));
            dsum += g * w2[ch];
            csum += g * w2[OC + ch];
        }
        dsum += __shfl_xor(dsum, 16, 32);                      // combine lane n / n+16
        csum += __shfl_xor(csum, 16, 32);
        int col = (ngrp * 6 + nt) * 16 + n;
        if (hi == 0) { sacc[0][mtile][col] = dsum; sacc[1][mtile][col] = csum; }
    }
    __syncthreads();
    for (int wc = tid; wc < WW; wc += 256) {
        float d = sacc[0][0][wc] + sacc[0][1][wc] + sacc[0][2][wc] + sacc[0][3][wc] + c2b0;
        float l = sacc[1][0][wc] + sacc[1][1][wc] + sacc[1][2][wc] + sacc[1][3][wc] + c2b1;
        size_t off = ((size_t)b * HH + h) * WW + wc;
        depth_out[off] = d;
        logit_out[off] = l;
    }
}

// ---------------------------------------------------------------------------
// 5) per-batch softmax over H*W (exp stored during sum pass, then rescaled)
// ---------------------------------------------------------------------------
__global__ void softmax_kernel(const float* __restrict__ logit, float* __restrict__ conf) {
    int b = blockIdx.x, tid = threadIdx.x;
    const float* L = logit + (size_t)b * HW;
    float* Cg = conf + (size_t)b * HW;
    __shared__ float red[256];
    float m = -3.4e38f;
    for (int i = tid; i < HW; i += 256) m = fmaxf(m, L[i]);
    red[tid] = m; __syncthreads();
    for (int o = 128; o > 0; o >>= 1) { if (tid < o) red[tid] = fmaxf(red[tid], red[tid + o]); __syncthreads(); }
    float mx = red[0]; __syncthreads();
    float s = 0.f;
    for (int i = tid; i < HW; i += 256) { float e = expf(L[i] - mx); Cg[i] = e; s += e; }
    red[tid] = s; __syncthreads();
    for (int o = 128; o > 0; o >>= 1) { if (tid < o) red[tid] += red[tid + o]; __syncthreads(); }
    float inv = 1.0f / red[0];
    for (int i = tid; i < HW; i += 256) Cg[i] *= inv;
}

// ---------------------------------------------------------------------------
// 6) 3x3 box sum / 9 (zero pad)
// ---------------------------------------------------------------------------
__global__ void box_kernel(const float* __restrict__ conf, float* __restrict__ out) {
    int blk = blockIdx.x; int b = blk / HH, h = blk % HH;
    int w = threadIdx.x; if (w >= WW) return;
    const float* Cb = conf + (size_t)b * HW;
    float s = 0.f;
    #pragma unroll
    for (int dh = -1; dh <= 1; ++dh) {
        int hh2 = h + dh; if (hh2 < 0 || hh2 >= HH) continue;
        #pragma unroll
        for (int dw = -1; dw <= 1; ++dw) {
            int ww2 = w + dw; if (ww2 < 0 || ww2 >= WW) continue;
            s += Cb[hh2 * WW + ww2];
        }
    }
    out[((size_t)b * HH + h) * WW + w] = s * (1.0f / 9.0f);
}

// ---------------------------------------------------------------------------
// 7) per-batch top-512: bit-level threshold binary search + 512-elem bitonic
//    sort in LDS (desc by conf, asc by index), emit (x, y, depth)
// ---------------------------------------------------------------------------
__global__ void topk_kernel(const float* __restrict__ conf, const float* __restrict__ depth,
                            float* __restrict__ pts) {
    int b = blockIdx.x, tid = threadIdx.x;
    const float* Cb = conf + (size_t)b * HW;
    const float* Db = depth + (size_t)b * HW;
    __shared__ int   red[256];
    __shared__ unsigned slo, shi;
    __shared__ int   scount;
    __shared__ float svals[KPTS];
    __shared__ int   sidx[KPTS];

    if (tid == 0) { slo = 0u; shi = 0x7f000000u; }
    __syncthreads();
    for (int it = 0; it < 33; ++it) {
        unsigned mid = (slo + shi) >> 1;
        float t = __uint_as_float(mid);
        int c = 0;
        for (int i = tid; i < HW; i += 256) c += (Cb[i] > t) ? 1 : 0;
        red[tid] = c; __syncthreads();
        for (int o = 128; o > 0; o >>= 1) { if (tid < o) red[tid] += red[tid + o]; __syncthreads(); }
        if (tid == 0) { if (red[0] >= KPTS) slo = mid; else shi = mid; }
        __syncthreads();
    }
    float thi = __uint_as_float(shi);   // count(conf > thi) < 512 <= count(conf >= thi)

    for (int e = tid; e < KPTS; e += 256) { svals[e] = -3.4e38f; sidx[e] = 0; }
    if (tid == 0) scount = 0;
    __syncthreads();
    for (int i = tid; i < HW; i += 256)
        if (Cb[i] > thi) { int p = atomicAdd(&scount, 1); if (p < KPTS) { svals[p] = Cb[i]; sidx[p] = i; } }
    __syncthreads();
    if (tid == 0) scount = min(scount, KPTS);
    __syncthreads();
    for (int i = tid; i < HW; i += 256)
        if (Cb[i] == thi) { int p = atomicAdd(&scount, 1); if (p < KPTS) { svals[p] = Cb[i]; sidx[p] = i; } }
    __syncthreads();

    for (int k = 2; k <= KPTS; k <<= 1) {
        for (int j = k >> 1; j > 0; j >>= 1) {
            for (int e = tid; e < KPTS; e += 256) {
                int p = e ^ j;
                if (p > e) {
                    bool up = ((e & k) == 0);
                    float ve = svals[e], vp = svals[p];
                    int   ie = sidx[e],  ip = sidx[p];
                    bool ePrec = (ve > vp) || (ve == vp && ie < ip);
                    bool pPrec = (vp > ve) || (vp == ve && ip < ie);
                    bool doswap = up ? pPrec : ePrec;
                    if (doswap) {
                        svals[e] = vp; svals[p] = ve;
                        sidx[e] = ip;  sidx[p] = ie;
                    }
                }
            }
            __syncthreads();
        }
    }
    for (int e = tid; e < KPTS; e += 256) {
        int idx2 = sidx[e];
        float x = (float)(idx2 % WW) * (1.0f / WW);
        float y = (float)(idx2 / WW) * (1.0f / HH);
        float* o = pts + ((size_t)b * KPTS + e) * 3;
        o[0] = x; o[1] = y; o[2] = Db[idx2];
    }
}

// ---------------------------------------------------------------------------
extern "C" void kernel_launch(void* const* d_in, const int* in_sizes, int n_in,
                              void* d_out, int out_size, void* d_ws, size_t ws_size,
                              hipStream_t stream) {
    const float* feat = (const float*)d_in[0];
    const float* aw   = (const float*)d_in[1];
    const float* ab   = (const float*)d_in[2];
    const float* w1   = (const float*)d_in[3];
    const float* b1   = (const float*)d_in[4];
    const float* w2   = (const float*)d_in[5];
    const float* b2   = (const float*)d_in[6];

    float* out   = (float*)d_out;
    float* pts   = out;                                // B*512*3
    float* depth = out + BATCH * KPTS * 3;             // B*H*W
    float* confo = depth + (size_t)BATCH * HW;         // B*H*W

    char*  ws    = (char*)d_ws;
    float* pool  = (float*)ws;                         // 2048 f
    float* attn  = pool + 2048;                        // 2048 f
    _Float16* wp = (_Float16*)(ws + 16384);            // 16 * 73728 halves (2,359,296 B)
    float* logit = (float*)(ws + 16384 + (size_t)BATCH * WPACK * 2);
    float* conf  = logit + (size_t)BATCH * HW;

    pool_kernel  <<<BATCH * CIN, 256, 0, stream>>>(feat, pool);
    attn_kernel  <<<8, 256, 0, stream>>>(pool, aw, ab, attn);
    pack_w_kernel<<<BATCH * 288, 256, 0, stream>>>(w1, attn, wp);
    conv_fused_kernel<<<BATCH * HH, 256, 0, stream>>>(feat, wp, b1, w2, b2, depth, logit);
    softmax_kernel<<<BATCH, 256, 0, stream>>>(logit, conf);
    box_kernel   <<<BATCH * HH, 192, 0, stream>>>(conf, confo);
    topk_kernel  <<<BATCH, 256, 0, stream>>>(confo, depth, pts);
}
